// Sae_17454747091439
// MI455X (gfx1250) — matrix operand order repaired and checked by reading, not compiled, not run on hardware
//
#include <hip/hip_runtime.h>

// ---------------------------------------------------------------------------
// SAE encoder forward, MI455X (gfx1250, wave32):
//   pre = (x - b_dec) @ W_enc^T + b_enc ; relu ; top-32 per token
// bf16 WMMA 16x16x32, f32 accum. W_enc pre-converted to bf16 in d_ws, then
// streamed into LDS with GLOBAL_LOAD_ASYNC_TO_LDS_B128 at prefetch depth 2
// (triple-buffered, s_wait_asynccnt 2). Top-k scan of the previous chunk is
// interleaved with WMMA compute (VALU/LDS co-executes with the XDL pipe).
// ---------------------------------------------------------------------------

#define D_IN    768
#define L_LAT   24576
#define KSEL    32
#define MTILE   64                 // tokens per workgroup
#define NTILE   128                // latents per chunk
#define KSTEP   32                 // bf16 WMMA K
#define NCHUNK  (L_LAT / NTILE)    // 192
#define KSTEPS  (D_IN / KSTEP)     // 24
#define TOTSTEP (NCHUNK * KSTEPS)  // 4608
#define THREADS 256                // 8 wave32 -> 2 waves/SIMD

#define ASTR 776                   // halves per A row (768 + 8 pad)
#define BSTR 40                    // halves per B row (32 + 8 pad)
#define BBUF 3                     // B tile triple buffer (async depth 2)
#define SSTR 132                   // floats per staging row (128 + 4 pad)
#define SCAN_PER_KS 6              // 24 * 6 = 144 >= 128 staged cols

typedef __bf16 v16bf __attribute__((ext_vector_type(16), aligned(16)));
typedef __bf16 v8bf  __attribute__((ext_vector_type(8)));
typedef __bf16 v4bf  __attribute__((ext_vector_type(4)));
typedef float  v8f   __attribute__((ext_vector_type(8)));
typedef int    v4i   __attribute__((ext_vector_type(4)));

typedef __attribute__((address_space(1))) v4i* gptr4;   // global int4*
typedef __attribute__((address_space(3))) v4i* lptr4;   // LDS int4*

#define W_BF16_BYTES ((size_t)L_LAT * D_IN * 2)

#define SMEM_BYTES ( (size_t)MTILE*ASTR*2            /* A tile bf16       */ \
                   + (size_t)BBUF*NTILE*BSTR*2       /* B triple buffer   */ \
                   + (size_t)2*MTILE*SSTR*4          /* staging x2 f32    */ \
                   + (size_t)D_IN*4                  /* b_dec copy        */ \
                   + (size_t)MTILE*KSEL*4            /* top-k values      */ \
                   + (size_t)MTILE*KSEL*4 )          /* top-k indices     */

// ---- gfx1250 async copy helpers -------------------------------------------
__device__ __forceinline__ void async_copy_b128(const void* g, void* l) {
#if __has_builtin(__builtin_amdgcn_global_load_async_to_lds_b128)
    __builtin_amdgcn_global_load_async_to_lds_b128((gptr4)g, (lptr4)l, 0, 0);
#else
    asm volatile("global_load_async_to_lds_b128 %0, %1, off"
                 :: "v"((unsigned)(size_t)l), "v"((unsigned long long)(size_t)g)
                 : "memory");
#endif
}
template <int N>
__device__ __forceinline__ void wait_async() {
#if __has_builtin(__builtin_amdgcn_s_wait_asynccnt)
    __builtin_amdgcn_s_wait_asynccnt(N);   // immediate required
#else
    asm volatile("s_wait_asynccnt %0" :: "i"(N) : "memory");
#endif
}

// ---- one-shot W_enc f32 -> bf16 pre-pass into d_ws ------------------------
__global__ __launch_bounds__(256)
void convert_w_kernel(const float* __restrict__ W, __bf16* __restrict__ Wb) {
    const size_t total = (size_t)L_LAT * D_IN / 8;   // 16B bf16 stores
    for (size_t q = (size_t)blockIdx.x * 256 + threadIdx.x; q < total;
         q += (size_t)gridDim.x * 256) {
        float4 a = *((const float4*)W + 2 * q);
        float4 b = *((const float4*)W + 2 * q + 1);
        v8bf pk;
        pk[0] = (__bf16)a.x; pk[1] = (__bf16)a.y; pk[2] = (__bf16)a.z; pk[3] = (__bf16)a.w;
        pk[4] = (__bf16)b.x; pk[5] = (__bf16)b.y; pk[6] = (__bf16)b.z; pk[7] = (__bf16)b.w;
        *((v8bf*)Wb + q) = pk;
    }
}

// ---------------------------------------------------------------------------
template <bool WSB>   // true: bf16 W from d_ws via async-LDS; false: f32 fallback
__global__ __launch_bounds__(THREADS, 1)
void sae_topk_kernel(const float* __restrict__ x,
                     const float* __restrict__ Wf,
                     const __bf16* __restrict__ Wb,
                     const float* __restrict__ benc_g,
                     const float* __restrict__ bdec_g,
                     float* __restrict__ out,
                     int Ntok)
{
    extern __shared__ char smem_raw[];
    __bf16* Ash  = (__bf16*)smem_raw;                       // [MTILE][ASTR]
    __bf16* Bsh  = Ash + MTILE * ASTR;                      // [BBUF][NTILE][BSTR]
    float*  stg  = (float*)(Bsh + BBUF * NTILE * BSTR);     // [2][MTILE][SSTR]
    float*  bdec = stg + 2 * MTILE * SSTR;                  // [D_IN]
    float*  topv = bdec + D_IN;                             // [MTILE][KSEL]
    int*    topi = (int*)(topv + MTILE * KSEL);             // [MTILE][KSEL]

    const int th   = threadIdx.x;
    const int lane = th & 31;
    const int wv   = th >> 5;            // 0..7
    const int lh   = lane >> 4;          // half-wave 0/1
    const int ln16 = lane & 15;
    const int mrow = wv >> 2;            // 0..1 : which 32-row half
    const int ncg  = (wv & 3) * 32;      // 32-col group
    const size_t tok0 = (size_t)blockIdx.x * MTILE;

    // ---- init -------------------------------------------------------------
    for (int i = th; i < D_IN; i += THREADS) bdec[i] = bdec_g[i];
    for (int i = th; i < MTILE * KSEL; i += THREADS) { topv[i] = -1.0f; topi[i] = 0; }
    __syncthreads();

    // ---- Phase 0: A = bf16(x - b_dec); batched loads to avoid serial waits -
    {
        const int PER = (MTILE * D_IN / 4) / THREADS;  // 48 float4 per thread
        float4 vr[8];
        for (int g = 0; g < PER; g += 8) {
            #pragma unroll
            for (int u = 0; u < 8; ++u) {
                int q = (g + u) * THREADS + th;
                int row = q / (D_IN / 4), c = q % (D_IN / 4);
                vr[u] = *((const float4*)(x + (tok0 + row) * D_IN) + c);
            }
            #pragma unroll
            for (int u = 0; u < 8; ++u) {
                int q = (g + u) * THREADS + th;
                int row = q / (D_IN / 4), c = q % (D_IN / 4);
                float4 bd = *((const float4*)bdec + c);
                v4bf pk;
                pk[0] = (__bf16)(vr[u].x - bd.x); pk[1] = (__bf16)(vr[u].y - bd.y);
                pk[2] = (__bf16)(vr[u].z - bd.z); pk[3] = (__bf16)(vr[u].w - bd.w);
                *(v4bf*)(Ash + row * ASTR + c * 4) = pk;
            }
        }
    }

    // ---- W tile pipeline ---------------------------------------------------
    float4 wreg[4];   // f32-fallback in-flight stage

    auto issueAsync = [&](int g) {       // stage g -> buffer g % BBUF
        int nc = g / KSTEPS, ks = g % KSTEPS;
        __bf16* Bb = Bsh + (g % BBUF) * NTILE * BSTR;
        #pragma unroll
        for (int i = 0; i < 2; ++i) {                 // 512 x 16B / 256 thr
            int q = i * THREADS + th;
            int row = q >> 2, f = q & 3;
            const __bf16* gp = Wb + ((size_t)nc * NTILE + row) * D_IN
                                  + ks * KSTEP + f * 8;
            async_copy_b128(gp, Bb + row * BSTR + f * 8);
        }
    };
    auto issueRegs = [&](int g) {        // f32 fallback: loads into registers
        int nc = g / KSTEPS, ks = g % KSTEPS;
        #pragma unroll
        for (int i = 0; i < 4; ++i) {                 // 1024 float4 / 256 thr
            int q = i * THREADS + th;
            int row = q >> 3, f = q & 7;
            wreg[i] = *(const float4*)(Wf + ((size_t)nc * NTILE + row) * D_IN
                                          + ks * KSTEP + f * 4);
        }
    };
    auto commitRegs = [&](int g) {       // f32 fallback: cvt + store stage g
        __bf16* Bb = Bsh + (g & 1) * NTILE * BSTR;
        #pragma unroll
        for (int i = 0; i < 4; ++i) {
            int q = i * THREADS + th;
            int row = q >> 3, f = q & 7;
            v4bf pk;
            pk[0] = (__bf16)wreg[i].x; pk[1] = (__bf16)wreg[i].y;
            pk[2] = (__bf16)wreg[i].z; pk[3] = (__bf16)wreg[i].w;
            *(v4bf*)(Bb + row * BSTR + f * 4) = pk;
        }
    };

    // top-k insertion (owner threads, LDS-resident lists, cached running min)
    float minv = -1.0f;
    int   minpos = 0;
    auto tryInsert = [&](float v, int gidx) {
        if (v > minv) {
            float* tv = topv + th * KSEL;
            topv[th * KSEL + minpos] = v;
            topi[th * KSEL + minpos] = gidx;
            minv = tv[0]; minpos = 0;
            #pragma unroll
            for (int i2 = 1; i2 < KSEL; ++i2) {
                float t2 = tv[i2];
                if (t2 < minv) { minv = t2; minpos = i2; }
            }
        }
    };

    // prologue: prime the pipeline
    if (WSB) { issueAsync(0); issueAsync(1); }
    else     { issueRegs(0); }

    const v8f vzero = {0.f, 0.f, 0.f, 0.f, 0.f, 0.f, 0.f, 0.f};

    for (int nc = 0; nc < NCHUNK; ++nc) {
        float* stgCur  = stg + (nc & 1) * MTILE * SSTR;
        float* stgPrev = stg + ((nc ^ 1) & 1) * MTILE * SSTR;

        v8f acc[2][2];
        #pragma unroll
        for (int mi = 0; mi < 2; ++mi)
            #pragma unroll
            for (int ct = 0; ct < 2; ++ct) acc[mi][ct] = vzero;

        for (int ks = 0; ks < KSTEPS; ++ks) {
            const int g = nc * KSTEPS + ks;
            int curbuf;
            if (WSB) {
                wait_async<2>();   // stage g done; stage g+1 may stay in flight
                __syncthreads();   // all waves' stage-g writes visible;
                                   // all waves done reading buffer (g+2)%BBUF
                if (g + 2 < TOTSTEP) issueAsync(g + 2);
                curbuf = g % BBUF;
            } else {
                commitRegs(g);     // cvt+store stage g (loads issued at g-1)
                __syncthreads();
                if (g + 1 < TOTSTEP) issueRegs(g + 1);
                curbuf = g & 1;
            }

            const __bf16* Bb = Bsh + curbuf * NTILE * BSTR;
            v16bf bfr[2];
            #pragma unroll
            for (int ct = 0; ct < 2; ++ct)
                bfr[ct] = *(const v16bf*)(Bb + (ncg + ct * 16 + ln16) * BSTR + lh * 16);
            #pragma unroll
            for (int mi = 0; mi < 2; ++mi) {
                int row = mrow * 32 + mi * 16 + ln16;
                const __bf16* Ap = Ash + row * ASTR + ks * KSTEP + lh * 16;
                v8bf alo = *(const v8bf*)Ap;
                v8bf ahi = *(const v8bf*)(Ap + 8);
                v16bf afr = __builtin_shufflevector(alo, ahi,
                    0,1,2,3,4,5,6,7,8,9,10,11,12,13,14,15);
                #pragma unroll
                for (int ct = 0; ct < 2; ++ct)
                    acc[mi][ct] = __builtin_amdgcn_wmma_f32_16x16x32_bf16(
                        false, afr, false, bfr[ct], (short)0, acc[mi][ct],
                        false, false);
            }

            // interleaved top-k scan of PREVIOUS chunk (co-executes with WMMA)
            if (nc > 0 && th < MTILE) {
                int j0 = ks * SCAN_PER_KS;
                int j1 = j0 + SCAN_PER_KS; if (j1 > NTILE) j1 = NTILE;
                const float* srow = stgPrev + th * SSTR;
                int base = (nc - 1) * NTILE;
                for (int j = j0; j < j1; ++j) tryInsert(srow[j], base + j);
            }
        }

        // ---- epilogue: +b_enc, ReLU, stage into current buffer ------------
        #pragma unroll
        for (int mi = 0; mi < 2; ++mi) {
            #pragma unroll
            for (int ct = 0; ct < 2; ++ct) {
                int col = ncg + ct * 16 + ln16;
                float be = benc_g[nc * NTILE + col];
                #pragma unroll
                for (int v = 0; v < 8; ++v) {
                    int row = mrow * 32 + mi * 16 + lh * 8 + v;
                    float val = acc[mi][ct][v] + be;
                    stgCur[row * SSTR + col] = val > 0.f ? val : 0.f;
                }
            }
        }
        // next chunk's first barrier makes stgCur visible before it is scanned
    }
    wait_async<0>();
    __syncthreads();

    // ---- drain: scan the last staged chunk --------------------------------
    if (th < MTILE) {
        const float* srow = stg + ((NCHUNK - 1) & 1) * MTILE * SSTR + th * SSTR;
        int base = (NCHUNK - 1) * NTILE;
        for (int j = 0; j < NTILE; ++j) tryInsert(srow[j], base + j);
    }
    __syncthreads();

    // ---- finalize: sort 32 entries descending, write acts + indices -------
    if (th < MTILE) {
        float* tv = topv + th * KSEL;
        int*   ti = topi + th * KSEL;
        for (int i = 0; i < KSEL; ++i) {
            int best = i; float bv = tv[i];
            for (int j = i + 1; j < KSEL; ++j) {
                float c = tv[j];
                if (c > bv) { bv = c; best = j; }
            }
            float t0 = tv[i]; tv[i] = tv[best]; tv[best] = t0;
            int   i0 = ti[i]; ti[i] = ti[best]; ti[best] = i0;
        }
        size_t token = tok0 + th;
        for (int i = 0; i < KSEL; ++i) {
            out[token * KSEL + i] = tv[i];
            out[(size_t)Ntok * KSEL + token * KSEL + i] = (float)ti[i];
        }
    }
}

extern "C" void kernel_launch(void* const* d_in, const int* in_sizes, int n_in,
                              void* d_out, int out_size, void* d_ws, size_t ws_size,
                              hipStream_t stream) {
    const float* x    = (const float*)d_in[0];
    const float* W    = (const float*)d_in[1];
    const float* benc = (const float*)d_in[2];
    const float* bdec = (const float*)d_in[3];
    float* out = (float*)d_out;

    int Ntok = in_sizes[0] / D_IN;
    int nblocks = Ntok / MTILE;

    if (ws_size >= W_BF16_BYTES) {
        __bf16* Wb = (__bf16*)d_ws;
        convert_w_kernel<<<2048, 256, 0, stream>>>(W, Wb);
        (void)hipFuncSetAttribute((const void*)sae_topk_kernel<true>,
                                  hipFuncAttributeMaxDynamicSharedMemorySize,
                                  (int)SMEM_BYTES);
        sae_topk_kernel<true><<<nblocks, THREADS, SMEM_BYTES, stream>>>(
            x, W, Wb, benc, bdec, out, Ntok);
    } else {
        (void)hipFuncSetAttribute((const void*)sae_topk_kernel<false>,
                                  hipFuncAttributeMaxDynamicSharedMemorySize,
                                  (int)SMEM_BYTES);
        sae_topk_kernel<false><<<nblocks, THREADS, SMEM_BYTES, stream>>>(
            x, W, (const __bf16*)nullptr, benc, bdec, out, Ntok);
    }
}
